// StateBayesianKalmanNet_5832565588449
// MI455X (gfx1250) — matrix operand order repaired and checked by reading, not compiled, not run on hardware
//
#include <hip/hip_runtime.h>

typedef __attribute__((ext_vector_type(16))) _Float16 v16h;
typedef __attribute__((ext_vector_type(8)))  _Float16 v8h;
typedef __attribute__((ext_vector_type(8)))  float    v8f;

constexpr int BDIM = 16384;
constexpr int JDIM = 10;
constexpr int SD   = 8;
constexpr int OD   = 4;
constexpr int HD   = 120;   // GRU hidden
constexpr int HP   = 128;   // padded hidden (4 WMMA K-steps of 32)
constexpr int OH   = 128;   // output-head hidden

// fragment-major weight regions (each fragment = 32 lanes * 16 halves = 512 halves)
constexpr int FR_IH  = 0;     // 3 gates * 8 nh * 4 kb = 96 frags
constexpr int FR_HH  = 96;    // 96 frags
constexpr int FR_FC2 = 192;   // 8 nt * 4 kb = 32 frags
constexpr int FR_FC3 = 224;   // 2 nt * 4 kb = 8 frags
constexpr int FR_TOT = 232;

__device__ __forceinline__ float sigm(float x) { return 1.f / (1.f + expf(-x)); }

__device__ __forceinline__ v8f wmma_f16(v16h a, v16h b, v8f c) {
    return __builtin_amdgcn_wmma_f32_16x16x32_f16(false, a, false, b, (short)0, c, false, false);
}

__device__ __forceinline__ void ld8(const float* __restrict__ p, float* r) {
    float4 a = ((const float4*)p)[0];
    float4 b = ((const float4*)p)[1];
    r[0] = a.x; r[1] = a.y; r[2] = a.z; r[3] = a.w;
    r[4] = b.x; r[5] = b.y; r[6] = b.z; r[7] = b.w;
}

// A-fragment (16-bit A 16x32 layout) straight from a global f32 row.
// Lane holds row m = lane&15; its 16 K-values are two contiguous 8-runs:
//   run0 = kb*32 + lh*8 .. +7   (always < 120)
//   run1 = run0 + 16            (== 120..127 only for kb==3,lh==1 -> zero pad)
__device__ __forceinline__ v16h frag_h_global(const float* __restrict__ row, int kb, int lh) {
    v16h a;
    float r[8];
    int c0 = kb * 32 + lh * 8;
    ld8(row + c0, r);
#pragma unroll
    for (int e = 0; e < 8; ++e) a[e] = (_Float16)r[e];
    int c1 = c0 + 16;
    if (c1 < HD) {
        ld8(row + c1, r);
#pragma unroll
        for (int e = 0; e < 8; ++e) a[8 + e] = (_Float16)r[e];
    } else {
#pragma unroll
        for (int e = 0; e < 8; ++e) a[8 + e] = (_Float16)0.f;
    }
    return a;
}

// Same, but a1 = z1[b,c] * dropout_mask(u1[j,b,c])
__device__ __forceinline__ v16h frag_a1_global(const float* __restrict__ zrow,
                                               const float* __restrict__ urow,
                                               int kb, int lh, float p1, float inv1) {
    v16h a;
    float z[8], u[8];
    int c0 = kb * 32 + lh * 8;
    ld8(zrow + c0, z); ld8(urow + c0, u);
#pragma unroll
    for (int e = 0; e < 8; ++e) a[e] = (_Float16)(z[e] * ((u[e] >= p1) ? inv1 : 0.f));
    int c1 = c0 + 16;
    if (c1 < HD) {
        ld8(zrow + c1, z); ld8(urow + c1, u);
#pragma unroll
        for (int e = 0; e < 8; ++e) a[8 + e] = (_Float16)(z[e] * ((u[e] >= p1) ? inv1 : 0.f));
    } else {
#pragma unroll
        for (int e = 0; e < 8; ++e) a[8 + e] = (_Float16)0.f;
    }
    return a;
}

// A-fragment from LDS stage [16 rows][128 cols] f16 (fully initialized): two 16B runs.
__device__ __forceinline__ v16h frag_a_lds(const _Float16* __restrict__ stg, int kb, int lane) {
    int lh = lane >> 4, m = lane & 15;
    const _Float16* row = stg + m * HP + kb * 32 + lh * 8;
    v8h lo = *(const v8h*)(row);
    v8h hi = *(const v8h*)(row + 16);
    return __builtin_shufflevector(lo, hi, 0, 1, 2, 3, 4, 5, 6, 7, 8, 9, 10, 11, 12, 13, 14, 15);
}

// ---------------- kernel 1: per-sample prediction + fc1 ----------------
__global__ void prep_kernel(const float* __restrict__ y_t, const float* __restrict__ xprev,
                            const float* __restrict__ dxp, const float* __restrict__ F,
                            const float* __restrict__ Hm, const float* __restrict__ Wfc1,
                            const float* __restrict__ bfc1,
                            float* __restrict__ z1, float* __restrict__ xp, float* __restrict__ inn) {
    int b = blockIdx.x;
    int t = threadIdx.x;
    __shared__ float sxp[SD], sinn[OD], snin[SD + OD];
    if (t < SD) {
        float a = 0.f;
        for (int u = 0; u < SD; ++u) a += xprev[b * SD + u] * F[t * SD + u];
        sxp[t] = a;
        xp[b * SD + t] = a;
    }
    __syncthreads();
    if (t < OD) {
        float a = 0.f;
        for (int s = 0; s < SD; ++s) a += sxp[s] * Hm[t * SD + s];
        float iv = y_t[b * OD + t] - a;
        sinn[t] = iv;
        inn[b * OD + t] = iv;
    }
    __syncthreads();
    if (t == 0) {
        float nd = 0.f;
        for (int u = 0; u < SD; ++u) { float v = dxp[b * SD + u]; nd += v * v; }
        nd = fmaxf(sqrtf(nd), 1e-12f);
        for (int u = 0; u < SD; ++u) snin[u] = dxp[b * SD + u] / nd;
        float ni = 0.f;
        for (int o = 0; o < OD; ++o) ni += sinn[o] * sinn[o];
        ni = fmaxf(sqrtf(ni), 1e-12f);
        for (int o = 0; o < OD; ++o) snin[SD + o] = sinn[o] / ni;
    }
    __syncthreads();
    if (t < HD) {
        float a = bfc1[t];
        for (int i = 0; i < SD + OD; ++i) a += Wfc1[t * (SD + OD) + i] * snin[i];
        z1[b * HD + t] = fmaxf(a, 0.f);
    }
}

// ---------------- kernel 2: fused GRU + heads, all GEMMs on WMMA ----------------
__global__ void __launch_bounds__(256) gru_kernel(
    const float* __restrict__ hprev, const float* __restrict__ Wih, const float* __restrict__ Whh,
    const float* __restrict__ bih, const float* __restrict__ bhh,
    const float* __restrict__ Wfc2, const float* __restrict__ bfc2,
    const float* __restrict__ Wfc3, const float* __restrict__ bfc3,
    const float* __restrict__ pl1, const float* __restrict__ pl2,
    const float* __restrict__ u1, const float* __restrict__ u2,
    const float* __restrict__ z1, const float* __restrict__ inn,
    float* __restrict__ corr) {
    extern __shared__ char smem[];
    _Float16* sFrag = (_Float16*)smem;                    // FR_TOT * 512 halves
    float*    sb    = (float*)(sFrag + FR_TOT * 512);     // 880 biases
    _Float16* stage = (_Float16*)(sb + 880);              // 8 waves * 16*128 f16

    int tid = threadIdx.x;
    // --- stage weights pre-swizzled into WMMA B-fragment layout (zero-padded) ---
    for (int idx = tid; idx < 96 * 512; idx += 256) {
        int f = idx >> 9, l = (idx >> 4) & 31, e = idx & 15;
        int g = f >> 5, rest = f & 31, nh = rest >> 2, kb = rest & 3;
        int n = nh * 16 + (l & 15);
        int k = kb * 32 + (l >> 4) * 8 + (e & 7) + (e >> 3) * 16;
        bool v = (n < HD) && (k < HD);
        sFrag[FR_IH * 512 + idx] = (_Float16)(v ? Wih[(g * HD + n) * HD + k] : 0.f);
        sFrag[FR_HH * 512 + idx] = (_Float16)(v ? Whh[(g * HD + n) * HD + k] : 0.f);
    }
    for (int idx = tid; idx < 32 * 512; idx += 256) {
        int f = idx >> 9, l = (idx >> 4) & 31, e = idx & 15;
        int nt = f >> 2, kb = f & 3;
        int n = nt * 16 + (l & 15);
        int k = kb * 32 + (l >> 4) * 8 + (e & 7) + (e >> 3) * 16;
        sFrag[FR_FC2 * 512 + idx] = (_Float16)((k < HD) ? Wfc2[n * HD + k] : 0.f);
    }
    for (int idx = tid; idx < 8 * 512; idx += 256) {
        int f = idx >> 9, l = (idx >> 4) & 31, e = idx & 15;
        int nt = f >> 2, kb = f & 3;
        int n = nt * 16 + (l & 15);
        int k = kb * 32 + (l >> 4) * 8 + (e & 7) + (e >> 3) * 16;
        sFrag[FR_FC3 * 512 + idx] = (_Float16)Wfc3[n * HP + k];
    }
    for (int i = tid; i < 360; i += 256) { sb[i] = bih[i]; sb[360 + i] = bhh[i]; }
    for (int i = tid; i < OH; i += 256) sb[720 + i] = bfc2[i];
    if (tid < 32) sb[848 + tid] = bfc3[tid];
    __syncthreads();

    const v16h* fragV = (const v16h*)sFrag;   // fragment chunk = [f*32 + lane]

    float p1 = sigm(pl1[0]), p2 = sigm(pl2[0]);
    float inv1 = 1.f / (1.f - p1), inv2 = 1.f / (1.f - p2);

    int wave = tid >> 5, lane = tid & 31;
    int lh = lane >> 4, nl = lane & 15, mrow = lane & 15;
    _Float16* myS = stage + wave * (16 * HP);
    float*    myF = (float*)myS;

    const int nTiles = JDIM * BDIM / 16;
    for (int tile = blockIdx.x * 8 + wave; tile < nTiles; tile += gridDim.x * 8) {
        int j  = tile >> 10;                    // BDIM/16 == 1024
        int b0 = (tile & 1023) << 4;
        size_t jb0 = (size_t)j * BDIM + b0;

        // --- A-fragments directly from global (no LDS round trip) ---
        const float* hrow = hprev + (jb0 + mrow) * HD;
        const float* zrow = z1 + (size_t)(b0 + mrow) * HD;
        const float* urow = u1 + (jb0 + mrow) * HD;
        v16h ah[4], aa[4];
#pragma unroll
        for (int kb = 0; kb < 4; ++kb) {
            ah[kb] = frag_h_global(hrow, kb, lh);
            aa[kb] = frag_a1_global(zrow, urow, kb, lh, p1, inv1);
        }

        // --- GRU: per 16-column block, 6 WMMA chains (r/z/n x ih/hh) ---
        for (int nh = 0; nh < 8; ++nh) {
            v8f xr = {}, xz = {}, xn = {}, hr = {}, hz = {}, hn = {};
#pragma unroll
            for (int kb = 0; kb < 4; ++kb) {
                int fb = nh * 4 + kb;
                xr = wmma_f16(aa[kb], fragV[(FR_IH + fb) * 32 + lane], xr);
                xz = wmma_f16(aa[kb], fragV[(FR_IH + 32 + fb) * 32 + lane], xz);
                xn = wmma_f16(aa[kb], fragV[(FR_IH + 64 + fb) * 32 + lane], xn);
                hr = wmma_f16(ah[kb], fragV[(FR_HH + fb) * 32 + lane], hr);
                hz = wmma_f16(ah[kb], fragV[(FR_HH + 32 + fb) * 32 + lane], hz);
                hn = wmma_f16(ah[kb], fragV[(FR_HH + 64 + fb) * 32 + lane], hn);
            }
            int col = nh * 16 + nl;
            bool cv = col < HD;
            float bir = cv ? sb[col] : 0.f,        biz = cv ? sb[120 + col] : 0.f;
            float bin_ = cv ? sb[240 + col] : 0.f, bhr = cv ? sb[360 + col] : 0.f;
            float bhz = cv ? sb[480 + col] : 0.f,  bhn = cv ? sb[600 + col] : 0.f;
#pragma unroll
            for (int i = 0; i < 8; ++i) {
                int m = lh * 8 + i;
                float hp = cv ? hprev[(jb0 + m) * HD + col] : 0.f;
                float r  = sigm(xr[i] + bir + hr[i] + bhr);
                float zz = sigm(xz[i] + biz + hz[i] + bhz);
                float nn = tanhf(xn[i] + bin_ + r * (hn[i] + bhn));
                float hv = (1.f - zz) * nn + zz * hp;
                myS[m * HP + col] = (_Float16)(cv ? hv : 0.f);
            }
        }
#pragma unroll
        for (int kb = 0; kb < 4; ++kb) ah[kb] = frag_a_lds(myS, kb, lane);  // h_new frags

        // --- a2 = relu(h_new @ Wfc2^T + b) * mask2 ---
        for (int nt = 0; nt < 8; ++nt) {
            v8f c = {};
#pragma unroll
            for (int kb = 0; kb < 4; ++kb)
                c = wmma_f16(ah[kb], fragV[(FR_FC2 + nt * 4 + kb) * 32 + lane], c);
            int col = nt * 16 + nl;
            float bb = sb[720 + col];
#pragma unroll
            for (int i = 0; i < 8; ++i) {
                int m = lh * 8 + i;
                float mk = (u2[(jb0 + m) * OH + col] >= p2) ? inv2 : 0.f;
                myS[m * HP + col] = (_Float16)(fmaxf(c[i] + bb, 0.f) * mk);
            }
        }
#pragma unroll
        for (int kb = 0; kb < 4; ++kb) aa[kb] = frag_a_lds(myS, kb, lane);  // a2 frags

        // --- K_vec = a2 @ Wfc3^T + b -> float staging [16][32] ---
        for (int nt = 0; nt < 2; ++nt) {
            v8f c = {};
#pragma unroll
            for (int kb = 0; kb < 4; ++kb)
                c = wmma_f16(aa[kb], fragV[(FR_FC3 + nt * 4 + kb) * 32 + lane], c);
            int col = nt * 16 + nl;
            float bb = sb[848 + col];
#pragma unroll
            for (int i = 0; i < 8; ++i) myF[(lh * 8 + i) * 32 + col] = c[i] + bb;
        }

        // --- correction[j,b,s] = sum_o K[m, s*4+o] * innovation[b,o] ---
#pragma unroll
        for (int q = 0; q < 4; ++q) {
            int idx = lane * 4 + q;
            int m = idx >> 3, s = idx & 7;
            float4 kv = *(const float4*)(myF + m * 32 + s * 4);
            float4 iv = *(const float4*)(inn + (size_t)(b0 + m) * OD);
            corr[(jb0 + m) * SD + s] = kv.x * iv.x + kv.y * iv.y + kv.z * iv.z + kv.w * iv.w;
        }
    }
}

// ---------------- kernel 3: mean over J + covariance ----------------
__global__ void reduce_kernel(const float* __restrict__ xp, const float* __restrict__ corr,
                              float* __restrict__ out) {
    int t = threadIdx.x;
    int b = blockIdx.x * 4 + (t >> 6);
    int e = t & 63, s = e >> 3, tt = e & 7;
    float cs[JDIM], ct[JDIM], ms = 0.f, mt = 0.f;
    for (int j = 0; j < JDIM; ++j) {
        cs[j] = corr[((size_t)j * BDIM + b) * SD + s];
        ct[j] = corr[((size_t)j * BDIM + b) * SD + tt];
        ms += cs[j]; mt += ct[j];
    }
    ms *= (1.f / JDIM); mt *= (1.f / JDIM);
    float p = 0.f;
    for (int j = 0; j < JDIM; ++j) p += (cs[j] - ms) * (ct[j] - mt);
    out[(size_t)BDIM * SD + (size_t)b * 64 + s * 8 + tt] = p * (1.f / JDIM);
    if (tt == 0) out[b * SD + s] = xp[b * SD + s] + ms;
}

// ---------------- kernel 4: concrete-dropout regularizer scalar ----------------
__global__ void reg_kernel(const float* __restrict__ Wfc1, const float* __restrict__ Wfc2,
                           const float* __restrict__ pl1, const float* __restrict__ pl2,
                           float* __restrict__ out) {
    __shared__ float r1[256], r2[256];
    int t = threadIdx.x;
    float s1 = 0.f, s2 = 0.f;
    for (int i = t; i < HD * (SD + OD); i += 256) { float v = Wfc1[i]; s1 += v * v; }
    for (int i = t; i < OH * HD; i += 256)        { float v = Wfc2[i]; s2 += v * v; }
    r1[t] = s1; r2[t] = s2;
    __syncthreads();
    for (int k = 128; k > 0; k >>= 1) {
        if (t < k) { r1[t] += r1[t + k]; r2[t] += r2[t + k]; }
        __syncthreads();
    }
    if (t == 0) {
        float p1 = sigm(pl1[0]), p2 = sigm(pl2[0]);
        float e1 = p1 * logf(p1) + (1.f - p1) * log1pf(-p1);
        float e2 = p2 * logf(p2) + (1.f - p2) * log1pf(-p2);
        out[(size_t)BDIM * SD + (size_t)BDIM * 64] =
            r1[0] / (1.f - p1) + e1 + r2[0] / (1.f - p2) + e2;
    }
}

extern "C" void kernel_launch(void* const* d_in, const int* in_sizes, int n_in,
                              void* d_out, int out_size, void* d_ws, size_t ws_size,
                              hipStream_t stream) {
    (void)in_sizes; (void)n_in; (void)out_size; (void)ws_size;
    const float* y_t   = (const float*)d_in[0];
    const float* xprev = (const float*)d_in[1];
    const float* dxp   = (const float*)d_in[2];
    const float* hprev = (const float*)d_in[3];
    const float* F     = (const float*)d_in[4];
    const float* Hm    = (const float*)d_in[5];
    const float* Wfc1  = (const float*)d_in[6];
    const float* bfc1  = (const float*)d_in[7];
    const float* Wih   = (const float*)d_in[8];
    const float* Whh   = (const float*)d_in[9];
    const float* bih   = (const float*)d_in[10];
    const float* bhh   = (const float*)d_in[11];
    const float* Wfc2  = (const float*)d_in[12];
    const float* bfc2  = (const float*)d_in[13];
    const float* Wfc3  = (const float*)d_in[14];
    const float* bfc3  = (const float*)d_in[15];
    const float* pl1   = (const float*)d_in[16];
    const float* pl2   = (const float*)d_in[17];
    const float* u1    = (const float*)d_in[18];
    const float* u2    = (const float*)d_in[19];
    float* out = (float*)d_out;

    float* ws   = (float*)d_ws;
    float* z1   = ws;                              // [B,120]
    float* xp   = z1 + (size_t)BDIM * HD;          // [B,8]
    float* inn  = xp + (size_t)BDIM * SD;          // [B,4]
    float* corr = inn + (size_t)BDIM * OD;         // [J,B,8]

    prep_kernel<<<BDIM, 128, 0, stream>>>(y_t, xprev, dxp, F, Hm, Wfc1, bfc1, z1, xp, inn);

    size_t lds = (size_t)FR_TOT * 512 * sizeof(_Float16)   // 237,568 B fragments
               + 880 * sizeof(float)                        //   3,520 B biases
               + (size_t)8 * 16 * HP * sizeof(_Float16);    //  32,768 B staging => 273,856 B
    gru_kernel<<<160, 256, lds, stream>>>(hprev, Wih, Whh, bih, bhh, Wfc2, bfc2, Wfc3, bfc3,
                                          pl1, pl2, u1, u2, z1, inn, corr);

    reduce_kernel<<<BDIM / 4, 256, 0, stream>>>(xp, corr, out);
    reg_kernel<<<1, 256, 0, stream>>>(Wfc1, Wfc2, pl1, pl2, out);
}
